// LocalAttention_1082331759401
// MI455X (gfx1250) — compile-verified, matrix operand order checked by reading, not run on hardware
//
#include <hip/hip_runtime.h>
#include <hip/hip_bf16.h>

// ---------------------------------------------------------------------------
// Local (banded) multi-head attention for MI455X (gfx1250), bf16 WMMA path.
// D = A x B + C with V_WMMA_F32_16X16X32_BF16 (f32 accumulate).
// Projection GEMMs use 4x4 register blocking (64x64 output per wave) for
// 64-way operand reuse; attention uses the 272-wide band (17 key tiles).
// ---------------------------------------------------------------------------

typedef __bf16 bf16_t;
typedef __attribute__((ext_vector_type(16))) __bf16 v16bf;
typedef __attribute__((ext_vector_type(8)))  __bf16 v8bf;
typedef __attribute__((ext_vector_type(8)))  float  v8f;

#define DEVFN static __device__ __forceinline__

constexpr int EMB   = 512;
constexpr int NSEQ  = 4096;
constexpr int BATCH = 2;
constexpr int HEADS = 8;
constexpr int HD    = 64;          // head dim
constexpr int WIN   = 128;         // attention window
constexpr int ROWS  = BATCH * NSEQ;      // 8192
constexpr float INV_SQRT_E = 0.044194173824159216f;  // 1/sqrt(512)

constexpr int NT   = 17;           // key tiles per query tile (272 keys)
constexpr int SROW = 276;          // f32 score row stride in LDS (272 + pad)
constexpr int PROW = 288;          // bf16 prob row stride (272 padded to 9*32)

// ---------------------------------------------------------------------------
// Fragment helpers (CDNA5 wave32 WMMA layouts, CDNA5 ISA §7.12.2)
// A (16x32 bf16): lane L holds row M=L%16; K runs at [kbase,+8) and
// [kbase+16,+8), kbase = (L>=16 ? 8 : 0). B loaded symmetric to A from the
// NxK (B^T) matrix. C/D f32: VGPR r -> (M=r + 8*(L>=16), N=L%16).
// ---------------------------------------------------------------------------

DEVFN v16bf load_frag(const bf16_t* row, int kofs) {
  v8bf lo = *(const v8bf*)(row + kofs);
  v8bf hi = *(const v8bf*)(row + kofs + 16);
  v16bf r;
#pragma unroll
  for (int i = 0; i < 8; ++i) { r[i] = lo[i]; r[i + 8] = hi[i]; }
  return r;
}

// Predicated B^T load along the key axis (8-aligned runs: fully in/out).
DEVFN v16bf load_bT_pred(const bf16_t* row, int jbase, int kofs) {
  v16bf r;
  const int j0 = jbase + kofs;
  const int j1 = j0 + 16;
  if (j0 >= 0 && j0 < NSEQ) {
    v8bf lo = *(const v8bf*)(row + j0);
#pragma unroll
    for (int i = 0; i < 8; ++i) r[i] = lo[i];
  } else {
#pragma unroll
    for (int i = 0; i < 8; ++i) r[i] = (__bf16)0.0f;
  }
  if (j1 >= 0 && j1 < NSEQ) {
    v8bf hi = *(const v8bf*)(row + j1);
#pragma unroll
    for (int i = 0; i < 8; ++i) r[i + 8] = hi[i];
  } else {
#pragma unroll
    for (int i = 0; i < 8; ++i) r[i + 8] = (__bf16)0.0f;
  }
  return r;
}

DEVFN v8f wmma_bf16(v16bf a, v16bf b, v8f c) {
  // (neg_a, A, neg_b, B, c_mod, C, reuse_a, reuse_b)
  return __builtin_amdgcn_wmma_f32_16x16x32_bf16(false, a, false, b,
                                                 (short)0, c, false, false);
}

// ---------------------------------------------------------------------------
// Kernel 0: f32 -> bf16 conversion
// ---------------------------------------------------------------------------
__global__ void k_cvt(const float* __restrict__ src, bf16_t* __restrict__ dst,
                      int n) {
  int i = blockIdx.x * blockDim.x + threadIdx.x;
  const int stride = gridDim.x * blockDim.x;
  for (; i < n; i += stride) dst[i] = (bf16_t)src[i];
}

// ---------------------------------------------------------------------------
// Kernel 1: fused QKV projection  Y = x @ W.T + b
//   4x4 register-blocked: each wave computes a 64x64 output block.
//   Q, K stored [b, h, n, d];  V stored transposed [b, h, d, n]
// ---------------------------------------------------------------------------
__global__ void __launch_bounds__(128) k_qkv(
    const bf16_t* __restrict__ xb,
    const bf16_t* __restrict__ Wq, const bf16_t* __restrict__ Wk,
    const bf16_t* __restrict__ Wv,
    const float* __restrict__ bq, const float* __restrict__ bk,
    const float* __restrict__ bv,
    bf16_t* __restrict__ Q, bf16_t* __restrict__ K, bf16_t* __restrict__ Vt) {
  const int lane  = threadIdx.x & 31;
  const int gw    = blockIdx.x * (blockDim.x >> 5) + (threadIdx.x >> 5);
  const int TILES = (ROWS / 64) * (EMB / 64);   // 64x64 blocks per matrix
  const int mat   = gw / TILES;                 // 0=Q 1=K 2=V (wave-uniform)
  const int rem   = gw % TILES;
  const int m0    = (rem / (EMB / 64)) * 64;
  const int c0    = (rem % (EMB / 64)) * 64;
  const int lrow  = lane & 15;
  const int kbase = (lane >> 4) * 8;
  const int mofs  = (lane >> 4) * 8;

  const bf16_t* W    = (mat == 0) ? Wq : (mat == 1) ? Wk : Wv;
  const float*  bias = (mat == 0) ? bq : (mat == 1) ? bk : bv;

  const bf16_t* arow[4];
  const bf16_t* brow[4];
#pragma unroll
  for (int i = 0; i < 4; ++i) {
    arow[i] = xb + (size_t)(m0 + i * 16 + lrow) * EMB;
    brow[i] = W  + (size_t)(c0 + i * 16 + lrow) * EMB;
  }

  v8f acc[4][4] = {};
#pragma unroll 2
  for (int kc = 0; kc < EMB / 32; ++kc) {
    const int ko = kc * 32 + kbase;
    v16bf a[4], b[4];
#pragma unroll
    for (int i = 0; i < 4; ++i) a[i] = load_frag(arow[i], ko);
#pragma unroll
    for (int j = 0; j < 4; ++j) b[j] = load_frag(brow[j], ko);
#pragma unroll
    for (int i = 0; i < 4; ++i)
#pragma unroll
      for (int j = 0; j < 4; ++j)
        acc[i][j] = wmma_bf16(a[i], b[j], acc[i][j]);
  }

#pragma unroll
  for (int j = 0; j < 4; ++j) {
    const int   c    = c0 + j * 16 + lrow;
    const float bval = bias[c];
    const int   h    = c >> 6, dd = c & 63;
#pragma unroll
    for (int i = 0; i < 4; ++i) {
#pragma unroll
      for (int r = 0; r < 8; ++r) {
        const int   M    = m0 + i * 16 + r + mofs;
        const int   bidx = M >> 12;          // /4096
        const int   nn   = M & (NSEQ - 1);
        const float val  = acc[i][j][r] + bval;
        if (mat == 0)
          Q[((size_t)(bidx * HEADS + h) * NSEQ + nn) * HD + dd] = (bf16_t)val;
        else if (mat == 1)
          K[((size_t)(bidx * HEADS + h) * NSEQ + nn) * HD + dd] = (bf16_t)val;
        else
          Vt[((size_t)(bidx * HEADS + h) * HD + dd) * NSEQ + nn] = (bf16_t)val;
      }
    }
  }
}

// ---------------------------------------------------------------------------
// Kernel 2: banded attention. One wave per (b, h, 16-row query tile).
//   Phase 1: S = Q . K^T for 17 key tiles (WMMA) -> LDS (f32)
//   Phase 2: masked softmax, 2 lanes per row, pair-reduce via __shfl_xor
//   Phase 3: O = P . V (P fragments hoisted to registers, reused for all
//            four d-tiles; V^T rows as predicated B operands)
// ---------------------------------------------------------------------------
__global__ void __launch_bounds__(64) k_attn(
    const bf16_t* __restrict__ Q, const bf16_t* __restrict__ K,
    const bf16_t* __restrict__ Vt, bf16_t* __restrict__ ctx) {
  __shared__ float  sS[2 * 16 * SROW];
  __shared__ __align__(16) bf16_t sP[2 * 16 * PROW];

  const int widx  = threadIdx.x >> 5;
  const int lane  = threadIdx.x & 31;
  const int gw    = blockIdx.x * 2 + widx;
  const int bh    = gw / (NSEQ / 16);       // batch*heads index, 0..15
  const int q0    = (gw % (NSEQ / 16)) * 16;
  const int lrow  = lane & 15;
  const int half  = lane >> 4;
  const int kbase = half * 8;
  const int mofs  = half * 8;

  float*  S = sS + widx * 16 * SROW;
  bf16_t* P = sP + widx * 16 * PROW;

  const bf16_t* Qb = Q  + (size_t)bh * NSEQ * HD;
  const bf16_t* Kb = K  + (size_t)bh * NSEQ * HD;
  const bf16_t* Vb = Vt + (size_t)bh * HD * NSEQ;

  // Q fragments, reused for all 17 key tiles
  const bf16_t* qrow = Qb + (size_t)(q0 + lrow) * HD;
  const v16bf aq0 = load_frag(qrow, kbase);
  const v16bf aq1 = load_frag(qrow, 32 + kbase);

  // ---- Phase 1: scores -------------------------------------------------
  for (int t = 0; t < NT; ++t) {
    const int kt0 = q0 - WIN + t * 16;
    if (kt0 < 0 || kt0 >= NSEQ) continue;   // wave-uniform: EXEC stays full
    const bf16_t* krow = Kb + (size_t)(kt0 + lrow) * HD;
    v8f acc = {};
    acc = wmma_bf16(aq0, load_frag(krow, kbase), acc);
    acc = wmma_bf16(aq1, load_frag(krow, 32 + kbase), acc);
#pragma unroll
    for (int r = 0; r < 8; ++r)
      S[(r + mofs) * SROW + t * 16 + lrow] = acc[r];
  }
  __syncthreads();

  // ---- Phase 2: masked softmax ----------------------------------------
  {
    const int   i    = q0 + lrow;
    float*      Srow = S + lrow * SROW;
    const int   k0   = half * 144;
    const int   k1   = k0 + 144;

    float m = -3.0e38f;
    for (int kk = k0; kk < k1 && kk < 272; ++kk) {
      const int j = q0 - WIN + kk;
      if (j >= 0 && j < NSEQ && (j - i) <= WIN && (i - j) <= WIN)
        m = fmaxf(m, Srow[kk]);
    }
    m = fmaxf(m, __shfl_xor(m, 16));

    float sum = 0.0f;
    for (int kk = k0; kk < k1 && kk < 272; ++kk) {
      const int  j  = q0 - WIN + kk;
      const bool ok = (j >= 0 && j < NSEQ && (j - i) <= WIN && (i - j) <= WIN);
      const float e = ok ? __expf((Srow[kk] - m) * INV_SQRT_E) : 0.0f;
      Srow[kk] = e;
      sum += e;
    }
    sum += __shfl_xor(sum, 16);
    const float inv = 1.0f / sum;

    bf16_t* Prow = P + lrow * PROW;
    for (int kk = k0; kk < k1; ++kk) {       // covers padded 0..287
      const float e = (kk < 272) ? Srow[kk] : 0.0f;
      Prow[kk] = (bf16_t)(e * inv);
    }
  }
  __syncthreads();

  // ---- Phase 3: O = P . V ---------------------------------------------
  // Hoist all 9 P fragments into registers; reuse across the 4 d-tiles.
  const bf16_t* prow = P + lrow * PROW;
  v16bf pf[PROW / 32];
#pragma unroll
  for (int c = 0; c < PROW / 32; ++c) pf[c] = load_frag(prow, c * 32 + kbase);

  const int b = bh >> 3, h = bh & 7;
#pragma unroll
  for (int dt = 0; dt < 4; ++dt) {
    const int d0 = dt * 16;
    const bf16_t* vrow = Vb + (size_t)(d0 + lrow) * NSEQ;
    v8f acc = {};
#pragma unroll
    for (int c = 0; c < PROW / 32; ++c) {    // 9 chunks of K=32
      v16bf bf = load_bT_pred(vrow, q0 - WIN, c * 32 + kbase);
      acc = wmma_bf16(pf[c], bf, acc);
    }
#pragma unroll
    for (int r = 0; r < 8; ++r) {
      const int nn = q0 + r + mofs;
      ctx[((size_t)b * NSEQ + nn) * EMB + h * HD + d0 + lrow] = (bf16_t)acc[r];
    }
  }
}

// ---------------------------------------------------------------------------
// Kernel 3: output projection  out = ctx @ Wo.T + bo  (f32 output)
//   4x4 register-blocked: each wave computes a 64x64 output block.
// ---------------------------------------------------------------------------
__global__ void __launch_bounds__(128) k_oproj(
    const bf16_t* __restrict__ ctx, const bf16_t* __restrict__ Wo,
    const float* __restrict__ bo, float* __restrict__ out) {
  const int lane  = threadIdx.x & 31;
  const int gw    = blockIdx.x * (blockDim.x >> 5) + (threadIdx.x >> 5);
  const int m0    = (gw / (EMB / 64)) * 64;
  const int c0    = (gw % (EMB / 64)) * 64;
  const int lrow  = lane & 15;
  const int kbase = (lane >> 4) * 8;
  const int mofs  = (lane >> 4) * 8;

  const bf16_t* arow[4];
  const bf16_t* brow[4];
#pragma unroll
  for (int i = 0; i < 4; ++i) {
    arow[i] = ctx + (size_t)(m0 + i * 16 + lrow) * EMB;
    brow[i] = Wo  + (size_t)(c0 + i * 16 + lrow) * EMB;
  }

  v8f acc[4][4] = {};
#pragma unroll 2
  for (int kc = 0; kc < EMB / 32; ++kc) {
    const int ko = kc * 32 + kbase;
    v16bf a[4], b[4];
#pragma unroll
    for (int i = 0; i < 4; ++i) a[i] = load_frag(arow[i], ko);
#pragma unroll
    for (int j = 0; j < 4; ++j) b[j] = load_frag(brow[j], ko);
#pragma unroll
    for (int i = 0; i < 4; ++i)
#pragma unroll
      for (int j = 0; j < 4; ++j)
        acc[i][j] = wmma_bf16(a[i], b[j], acc[i][j]);
  }

#pragma unroll
  for (int j = 0; j < 4; ++j) {
    const float bval = bo[c0 + j * 16 + lrow];
#pragma unroll
    for (int i = 0; i < 4; ++i) {
#pragma unroll
      for (int r = 0; r < 8; ++r)
        out[(size_t)(m0 + i * 16 + r + mofs) * EMB + c0 + j * 16 + lrow] =
            acc[i][j][r] + bval;
    }
  }
}

// ---------------------------------------------------------------------------
extern "C" void kernel_launch(void* const* d_in, const int* in_sizes, int n_in,
                              void* d_out, int out_size, void* d_ws,
                              size_t ws_size, hipStream_t stream) {
  const float* x  = (const float*)d_in[0];
  const float* Wq = (const float*)d_in[1];
  const float* bq = (const float*)d_in[2];
  const float* Wk = (const float*)d_in[3];
  const float* bk = (const float*)d_in[4];
  const float* Wv = (const float*)d_in[5];
  const float* bv = (const float*)d_in[6];
  const float* Wo = (const float*)d_in[7];
  const float* bo = (const float*)d_in[8];
  float* out = (float*)d_out;

  char*  ws  = (char*)d_ws;
  size_t off = 0;
  auto take = [&](size_t bytes) {
    char* p = ws + off;
    off += (bytes + 255) & ~(size_t)255;
    return p;
  };
  bf16_t* xb  = (bf16_t*)take((size_t)ROWS * EMB * 2);
  bf16_t* Wqb = (bf16_t*)take((size_t)EMB * EMB * 2);
  bf16_t* Wkb = (bf16_t*)take((size_t)EMB * EMB * 2);
  bf16_t* Wvb = (bf16_t*)take((size_t)EMB * EMB * 2);
  bf16_t* Wob = (bf16_t*)take((size_t)EMB * EMB * 2);
  bf16_t* Qb  = (bf16_t*)take((size_t)ROWS * EMB * 2);
  bf16_t* Kb  = (bf16_t*)take((size_t)ROWS * EMB * 2);
  bf16_t* Vtb = (bf16_t*)take((size_t)ROWS * EMB * 2);
  bf16_t* ctx = (bf16_t*)take((size_t)ROWS * EMB * 2);

  // f32 -> bf16
  k_cvt<<<1024, 256, 0, stream>>>(x,  xb,  ROWS * EMB);
  k_cvt<<<256,  256, 0, stream>>>(Wq, Wqb, EMB * EMB);
  k_cvt<<<256,  256, 0, stream>>>(Wk, Wkb, EMB * EMB);
  k_cvt<<<256,  256, 0, stream>>>(Wv, Wvb, EMB * EMB);
  k_cvt<<<256,  256, 0, stream>>>(Wo, Wob, EMB * EMB);

  // QKV projections: 3 * (8192/64) * (512/64) = 3072 waves, 4 waves/block
  k_qkv<<<768, 128, 0, stream>>>(xb, Wqb, Wkb, Wvb, bq, bk, bv, Qb, Kb, Vtb);

  // Banded attention: 2*8*256 = 4096 waves, 2 waves/block
  k_attn<<<2048, 64, 0, stream>>>(Qb, Kb, Vtb, ctx);

  // Output projection: (8192/64)*(512/64) = 1024 waves, 4 waves/block
  k_oproj<<<256, 128, 0, stream>>>(ctx, Wob, bo, out);
}
